// ClassNetPP_55731495633087
// MI455X (gfx1250) — matrix operand order repaired and checked by reading, not compiled, not run on hardware
//
#include <hip/hip_runtime.h>
#include <hip/hip_bf16.h>

// ---------------------------------------------------------------------------
// ClassNetPP forward on gfx1250. All GEMM stages: v_wmma_f32_16x16x32_f16
// (f16 in, f32 accumulate), software-pipelined K-loops. The two 3x3 convs are
// fused into one kernel that stages a column-halo-padded image slab in LDS via
// per-row Tensor Data Mover descriptors (tensor_load_to_lds + s_wait_tensorcnt)
// and feeds WMMA from unpredicated ds_load_b128 (no exec masking / no hazard
// NOPs in the inner loop).
//   B=16, N=1024 (32x32), DIN=768, D=512, HID=128, C=20, K=64
// ---------------------------------------------------------------------------

typedef __attribute__((ext_vector_type(16))) _Float16 v16h;
typedef __attribute__((ext_vector_type(8)))  _Float16 v8h;
typedef __attribute__((ext_vector_type(8)))  float    v8f;
typedef __attribute__((ext_vector_type(4)))  unsigned int u32x4;
typedef __attribute__((ext_vector_type(8)))  unsigned int u32x8;

#define BN_EPS 1e-5f

// --- WMMA helpers -----------------------------------------------------------

static __device__ inline v8f wmma_f16(v16h a, v16h b, v8f c) {
  return __builtin_amdgcn_wmma_f32_16x16x32_f16(
      false, a, false, b, (short)0, c, false, false);
}

// A-fragment (16x32 f16): lane<16: K 0..7 / 16..23; lane>=16: K 8..15 / 24..31
static __device__ inline v16h load_a16(const _Float16* __restrict__ row,
                                       int k0, int hi) {
  v8h lo = *(const v8h*)(row + k0 + hi * 8);
  v8h hh = *(const v8h*)(row + k0 + 16 + hi * 8);
  return __builtin_shufflevector(lo, hh,
      0,1,2,3,4,5,6,7,8,9,10,11,12,13,14,15);
}
// Same layout sourcing from LDS (-> ds_load_b128)
static __device__ inline v16h load_a16_lds(const _Float16* row, int k0, int hi) {
  v8h lo = *(const v8h*)(row + k0 + hi * 8);
  v8h hh = *(const v8h*)(row + k0 + 16 + hi * 8);
  return __builtin_shufflevector(lo, hh,
      0,1,2,3,4,5,6,7,8,9,10,11,12,13,14,15);
}

// B-fragment (32x16 f16): lane owns col n = lane&15; K 0..7 / 16..23 (lane<16), 8..15 / 24..31 (lane>=16).
static __device__ inline v16h load_b16(const _Float16* __restrict__ col,
                                       int k0, int hi) {
  v8h lo = *(const v8h*)(col + k0 + hi * 8);
  v8h hh = *(const v8h*)(col + k0 + 16 + hi * 8);
  return __builtin_shufflevector(lo, hh,
      0,1,2,3,4,5,6,7,8,9,10,11,12,13,14,15);
}

// Epilogues: 0 = BN+ReLU, 1 = BN + residual + ReLU, 2 = +bias (no ReLU)
template <int EPI>
static __device__ inline void store_frag(
    const v8f& c, int tmGlob, int tn, int rm, int cn, int li, int hi,
    const float* __restrict__ p0, const float* __restrict__ p1,
    const float* __restrict__ p2, const float* __restrict__ p3,
    const _Float16* __restrict__ resid, int residLd,
    _Float16* __restrict__ out, int outLd, int outColOff) {
  const int ch = tn + cn * 16 + li;
  float s = 1.0f, t = 0.0f;
  if (EPI == 0 || EPI == 1) {
    const float inv = rsqrtf(p3[ch] + BN_EPS);
    s = p0[ch] * inv;
    t = p1[ch] - p2[ch] * s;
  } else {
    t = p0[ch];
  }
  const int m0 = tmGlob + rm * 16 + hi * 8;
#pragma unroll
  for (int r = 0; r < 8; ++r) {
    float v = c[r] * s + t;
    if (EPI == 1) v += (float)resid[(size_t)(m0 + r) * residLd + ch];
    if (EPI != 2) v = fmaxf(v, 0.0f);
    out[(size_t)(m0 + r) * outLd + outColOff + ch] = (_Float16)v;
  }
}

// --- Generic f16 WMMA GEMM, software-pipelined ------------------------------
// One wave owns a 32x64 output tile (2 row-frags x 4 col-frags); fragments for
// k+32 load while wmma consumes k.
template <int K, int EPI>
__global__ __launch_bounds__(256) void gemm16(
    const _Float16* __restrict__ A, const _Float16* __restrict__ Bw,
    const float* __restrict__ p0, const float* __restrict__ p1,
    const float* __restrict__ p2, const float* __restrict__ p3,
    const _Float16* __restrict__ resid, _Float16* __restrict__ out,
    int M, int N, int outLd, int outColOff) {
  const int wave = (int)((blockIdx.x * blockDim.x + threadIdx.x) >> 5);
  const int lane = (int)(threadIdx.x & 31);
  const int li = lane & 15, hi = lane >> 4;
  const int tilesN = N >> 6;
  const int tm = (wave / tilesN) << 5;
  const int tn = (wave % tilesN) << 6;
  if (tm >= M) return;

  const _Float16* Ar0 = A + (size_t)(tm + li) * K;
  const _Float16* Ar1 = A + (size_t)(tm + 16 + li) * K;
  const _Float16* Bc0 = Bw + (size_t)(tn + li) * K;
  const _Float16* Bc1 = Bw + (size_t)(tn + 16 + li) * K;
  const _Float16* Bc2 = Bw + (size_t)(tn + 32 + li) * K;
  const _Float16* Bc3 = Bw + (size_t)(tn + 48 + li) * K;

  v8f acc[8] = {};
  v16h a0 = load_a16(Ar0, 0, hi), a1 = load_a16(Ar1, 0, hi);
  v16h b0 = load_b16(Bc0, 0, hi), b1 = load_b16(Bc1, 0, hi);
  v16h b2 = load_b16(Bc2, 0, hi), b3 = load_b16(Bc3, 0, hi);
#pragma unroll 2
  for (int k0 = 32; k0 < K; k0 += 32) {
    __builtin_prefetch(Ar0 + k0 + 32, 0, 1);
    __builtin_prefetch(Bc0 + k0 + 32, 0, 1);
    v16h na0 = load_a16(Ar0, k0, hi), na1 = load_a16(Ar1, k0, hi);
    v16h nb0 = load_b16(Bc0, k0, hi), nb1 = load_b16(Bc1, k0, hi);
    v16h nb2 = load_b16(Bc2, k0, hi), nb3 = load_b16(Bc3, k0, hi);
    acc[0] = wmma_f16(a0, b0, acc[0]);
    acc[1] = wmma_f16(a0, b1, acc[1]);
    acc[2] = wmma_f16(a0, b2, acc[2]);
    acc[3] = wmma_f16(a0, b3, acc[3]);
    acc[4] = wmma_f16(a1, b0, acc[4]);
    acc[5] = wmma_f16(a1, b1, acc[5]);
    acc[6] = wmma_f16(a1, b2, acc[6]);
    acc[7] = wmma_f16(a1, b3, acc[7]);
    a0 = na0; a1 = na1; b0 = nb0; b1 = nb1; b2 = nb2; b3 = nb3;
  }
  acc[0] = wmma_f16(a0, b0, acc[0]);
  acc[1] = wmma_f16(a0, b1, acc[1]);
  acc[2] = wmma_f16(a0, b2, acc[2]);
  acc[3] = wmma_f16(a0, b3, acc[3]);
  acc[4] = wmma_f16(a1, b0, acc[4]);
  acc[5] = wmma_f16(a1, b1, acc[5]);
  acc[6] = wmma_f16(a1, b2, acc[6]);
  acc[7] = wmma_f16(a1, b3, acc[7]);

#pragma unroll
  for (int rm = 0; rm < 2; ++rm)
#pragma unroll
    for (int cn = 0; cn < 4; ++cn)
      store_frag<EPI>(acc[rm * 4 + cn], tm, tn, rm, cn, li, hi,
                      p0, p1, p2, p3, resid, outLd, out, outLd, outColOff);
}

// --- Fused 3x3 convs (dil=1 and dil=2) from an LDS-staged padded slab -------
// Grid: 64 blocks = 16 images x 4 row-quarters. Block = 512 threads (16 waves).
// LDS slab: 12 rows x 36 px (2+32+2 col halo) x 128 ch f16 = 108 KB.
// Waves 0..11 each fill one LDS row: TDM descriptor for the 8192B in-image
// interior (or full zero-fill for out-of-image rows) + zeroed column halos.
// All tap reads are then unconditionally in-bounds: no exec-masked ds_loads.
__global__ __launch_bounds__(512) void conv_fused(
    const _Float16* __restrict__ Ain,   // [16*1024, 128]
    const _Float16* __restrict__ Wl,    // [9][128][128] (dil=1)
    const _Float16* __restrict__ Wg,    // [9][128][128] (dil=2)
    const float* __restrict__ gl, const float* __restrict__ bl,
    const float* __restrict__ ml, const float* __restrict__ vl,
    const float* __restrict__ gg, const float* __restrict__ bg,
    const float* __restrict__ mg, const float* __restrict__ vg,
    _Float16* __restrict__ out) {       // [16*1024, 256]
  extern __shared__ _Float16 simg[];    // [12][36][128]
  const int bimg = (int)blockIdx.x >> 2;
  const int quad = (int)blockIdx.x & 3;
  const int base = quad * 8;
  const _Float16* src = Ain + (size_t)bimg * 1024 * 128;
  const int wv = (int)(threadIdx.x >> 5);
  const int lane = (int)(threadIdx.x & 31);

  if (wv < 12) {                        // one LDS row per wave
    const int gh = base - 2 + wv;
    uint4* rowq = (uint4*)simg + (size_t)wv * 576;  // 576 uint4 = 9216 B/row
    uint4 z; z.x = z.y = z.z = z.w = 0u;
    if (gh >= 0 && gh < 32) {
      rowq[lane] = z;                   // left  halo: 32 uint4 = 512 B
      rowq[544 + lane] = z;             // right halo: 32 uint4 = 512 B
      // TDM: 8192 B (1024 x 8B) from image row gh -> LDS interior
      const unsigned long long ga =
          (unsigned long long)(uintptr_t)(src + (size_t)gh * 4096);
      const unsigned ldsOff = (unsigned)(wv * 9216 + 512);
      u32x4 g0;
      g0[0] = 1u;                                   // count=1
      g0[1] = ldsOff;                               // lds_addr (bytes)
      g0[2] = (unsigned)ga;                         // global_addr[31:0]
      g0[3] = ((unsigned)(ga >> 32) & 0x01FFFFFFu)  // global_addr[56:32]
              | (2u << 30);                         // type=2
      u32x8 g1;
      g1[0] = 3u << 16;                             // data_size=8B
      g1[1] = 1024u << 16;                          // tensor_dim0 = 1024
      g1[2] = 1u << 16;                             // tensor_dim1 = 1
      g1[3] = 1024u << 16;                          // tile_dim0 = 1024
      g1[4] = 0u;                                   // tile_dim1/2 unused
      g1[5] = 1024u;                                // tensor_dim0_stride
      g1[6] = 0u;
      g1[7] = 0u;
      asm volatile("tensor_load_to_lds %0, %1" :: "s"(g0), "s"(g1) : "memory");
      __builtin_amdgcn_s_wait_tensorcnt(0);
    } else {
      // whole row (incl. halos) is outside the image -> zero 576 uint4
#pragma unroll
      for (int i = 0; i < 18; ++i) rowq[lane + i * 32] = z;
    }
  }
  __syncthreads();

  const int li = lane & 15, hi = lane >> 4;

  // 64 tiles: {conv: l/g} x {8 pixel tiles of 32} x {4 col tiles of 32}
#pragma unroll
  for (int j = 0; j < 4; ++j) {
    const int t = wv + 16 * j;
    const int conv = t >> 5;            // 0 = local(dil1), 1 = global(dil2)
    const int rem = t & 31;
    const int tn = (rem & 3) << 5;
    const int ptl = rem >> 2;           // 0..7 local pixel tile
    const int dil = 1 + conv;
    const _Float16* W = conv ? Wg : Wl;

    const int lp0 = ptl * 32 + li, lp1 = lp0 + 16;
    const int hl0 = lp0 >> 5, w0 = lp0 & 31;
    const int hl1 = lp1 >> 5, w1 = lp1 & 31;

    v8f c00 = {}, c01 = {}, c10 = {}, c11 = {};
#pragma unroll
    for (int tap = 0; tap < 9; ++tap) {
      const int dy = (tap / 3 - 1) * dil, dx = (tap % 3 - 1) * dil;
      // padded slab: row idx hl+2+dy in [0,12), col idx w+dx+2 in [0,36)
      const _Float16* r0 = simg + (size_t)((hl0 + 2 + dy) * 36 + w0 + dx + 2) * 128;
      const _Float16* r1 = simg + (size_t)((hl1 + 2 + dy) * 36 + w1 + dx + 2) * 128;
      const _Float16* Bc0 = W + (size_t)tap * 128 * 128 + (size_t)(tn + li) * 128;
      const _Float16* Bc1 = W + (size_t)tap * 128 * 128 + (size_t)(tn + 16 + li) * 128;
#pragma unroll
      for (int k0 = 0; k0 < 128; k0 += 32) {
        v16h a0 = load_a16_lds(r0, k0, hi);
        v16h a1 = load_a16_lds(r1, k0, hi);
        v16h b0 = load_b16(Bc0, k0, hi);
        v16h b1 = load_b16(Bc1, k0, hi);
        c00 = wmma_f16(a0, b0, c00);
        c01 = wmma_f16(a0, b1, c01);
        c10 = wmma_f16(a1, b0, c10);
        c11 = wmma_f16(a1, b1, c11);
      }
    }
    const float* pg = conv ? gg : gl;
    const float* pb = conv ? bg : bl;
    const float* pm = conv ? mg : ml;
    const float* pv = conv ? vg : vl;
    const int tmGlob = bimg * 1024 + base * 32 + ptl * 32;
    const int colOff = conv << 7;
    store_frag<0>(c00, tmGlob, tn, 0, 0, li, hi, pg, pb, pm, pv, nullptr, 0, out, 256, colOff);
    store_frag<0>(c01, tmGlob, tn, 0, 1, li, hi, pg, pb, pm, pv, nullptr, 0, out, 256, colOff);
    store_frag<0>(c10, tmGlob, tn, 1, 0, li, hi, pg, pb, pm, pv, nullptr, 0, out, 256, colOff);
    store_frag<0>(c11, tmGlob, tn, 1, 1, li, hi, pg, pb, pm, pv, nullptr, 0, out, 256, colOff);
  }
}

// --- L2-normalize rows of Y[16384,512]; write f32 x-out and f16 copy -------
__global__ __launch_bounds__(128) void l2norm_k(
    const _Float16* __restrict__ Y, float* __restrict__ Xout,
    _Float16* __restrict__ Xn16) {
  const int row = (int)blockIdx.x;
  const int t = (int)threadIdx.x;
  const _Float16* yr = Y + (size_t)row * 512;
  float v[4], ss = 0.0f;
#pragma unroll
  for (int i = 0; i < 4; ++i) {
    v[i] = (float)yr[t + i * 128];
    ss += v[i] * v[i];
  }
#pragma unroll
  for (int off = 1; off < 32; off <<= 1) ss += __shfl_xor(ss, off, 32);
  __shared__ float part[4];
  if ((t & 31) == 0) part[t >> 5] = ss;
  __syncthreads();
  const float tot = part[0] + part[1] + part[2] + part[3];
  const float inv = 1.0f / fmaxf(sqrtf(tot), 1e-12f);
#pragma unroll
  for (int i = 0; i < 4; ++i) {
    const float o = v[i] * inv;
    Xout[(size_t)row * 512 + t + i * 128] = o;
    Xn16[(size_t)row * 512 + t + i * 128] = (_Float16)o;
  }
}

// --- sims: per (b,c,16-pixel tile) wave: GEMM 16x512x64 + max over 64 ------
__global__ __launch_bounds__(256) void sims_max_k(
    const _Float16* __restrict__ Xn, const _Float16* __restrict__ P,
    float* __restrict__ act) {
  const int wave = (int)((blockIdx.x * blockDim.x + threadIdx.x) >> 5);
  const int lane = (int)(threadIdx.x & 31);
  const int li = lane & 15, hi = lane >> 4;
  const int pt = wave & 63;
  const int c = (wave >> 6) % 20;
  const int b = wave / (64 * 20);

  const _Float16* Ar = Xn + (size_t)(b * 1024 + pt * 16 + li) * 512;
  const _Float16* Pc = P + (size_t)(c * 64 + li) * 512;

  v8f c0 = {}, c1 = {}, c2 = {}, c3 = {};
  v16h a = load_a16(Ar, 0, hi);
  v16h b0 = load_b16(Pc + 0 * 16 * 512, 0, hi);
  v16h b1 = load_b16(Pc + 1 * 16 * 512, 0, hi);
  v16h b2 = load_b16(Pc + 2 * 16 * 512, 0, hi);
  v16h b3 = load_b16(Pc + 3 * 16 * 512, 0, hi);
#pragma unroll 2
  for (int k0 = 32; k0 < 512; k0 += 32) {
    __builtin_prefetch(Ar + k0 + 32, 0, 1);
    v16h na = load_a16(Ar, k0, hi);
    v16h nb0 = load_b16(Pc + 0 * 16 * 512, k0, hi);
    v16h nb1 = load_b16(Pc + 1 * 16 * 512, k0, hi);
    v16h nb2 = load_b16(Pc + 2 * 16 * 512, k0, hi);
    v16h nb3 = load_b16(Pc + 3 * 16 * 512, k0, hi);
    c0 = wmma_f16(a, b0, c0);
    c1 = wmma_f16(a, b1, c1);
    c2 = wmma_f16(a, b2, c2);
    c3 = wmma_f16(a, b3, c3);
    a = na; b0 = nb0; b1 = nb1; b2 = nb2; b3 = nb3;
  }
  c0 = wmma_f16(a, b0, c0);
  c1 = wmma_f16(a, b1, c1);
  c2 = wmma_f16(a, b2, c2);
  c3 = wmma_f16(a, b3, c3);

  float mx[8];
#pragma unroll
  for (int r = 0; r < 8; ++r)
    mx[r] = fmaxf(fmaxf(c0[r], c1[r]), fmaxf(c2[r], c3[r]));
#pragma unroll
  for (int off = 1; off < 16; off <<= 1)
#pragma unroll
    for (int r = 0; r < 8; ++r) mx[r] = fmaxf(mx[r], __shfl_xor(mx[r], off, 32));
  if (li == 0) {
    const size_t basei = ((size_t)b * 20 + c) * 1024 + pt * 16 + hi * 8;
#pragma unroll
    for (int r = 0; r < 8; ++r) act[basei + r] = mx[r];
  }
}

// --- logits: max over 1024 pixels * logit_scale ----------------------------
__global__ __launch_bounds__(256) void logits_k(
    const float* __restrict__ act, const float* __restrict__ scale,
    float* __restrict__ logits) {
  const int bc = (int)blockIdx.x;
  const int t = (int)threadIdx.x;
  const float* ar = act + (size_t)bc * 1024;
  float m = fmaxf(fmaxf(ar[t], ar[t + 256]), fmaxf(ar[t + 512], ar[t + 768]));
#pragma unroll
  for (int off = 1; off < 32; off <<= 1) m = fmaxf(m, __shfl_xor(m, off, 32));
  __shared__ float p[8];
  if ((t & 31) == 0) p[t >> 5] = m;
  __syncthreads();
  if (t == 0) {
    float mm = p[0];
#pragma unroll
    for (int i = 1; i < 8; ++i) mm = fmaxf(mm, p[i]);
    logits[bc] = mm * scale[0];
  }
}

// --- format conversion / weight repack -------------------------------------
__global__ void cvt_f16(const float* __restrict__ s, _Float16* __restrict__ d,
                        int n) {
  int i = (int)(blockIdx.x * blockDim.x + threadIdx.x);
  if (i < n) d[i] = (_Float16)s[i];
}

// src: [O=128][I=128][3][3] f32 -> dst: [tap][O][I] f16
__global__ void repack3x3(const float* __restrict__ src,
                          _Float16* __restrict__ dst) {
  int i = (int)(blockIdx.x * blockDim.x + threadIdx.x);
  if (i >= 9 * 128 * 128) return;
  const int t = i / (128 * 128);
  const int rem = i % (128 * 128);
  const int o = rem / 128, ii = rem % 128;
  dst[i] = (_Float16)src[(size_t)(o * 128 + ii) * 9 + t];
}

// ---------------------------------------------------------------------------
extern "C" void kernel_launch(void* const* d_in, const int* in_sizes, int n_in,
                              void* d_out, int out_size, void* d_ws,
                              size_t ws_size, hipStream_t stream) {
  (void)in_sizes; (void)n_in; (void)out_size; (void)ws_size;
  const float* patch = (const float*)d_in[0];
  const float* protos = (const float*)d_in[1];
  const float* w0 = (const float*)d_in[2];
  const float* g0 = (const float*)d_in[3];
  const float* b0 = (const float*)d_in[4];
  const float* m0 = (const float*)d_in[5];
  const float* v0 = (const float*)d_in[6];
  const float* wr = (const float*)d_in[7];
  const float* gr = (const float*)d_in[8];
  const float* br = (const float*)d_in[9];
  const float* mr = (const float*)d_in[10];
  const float* vr = (const float*)d_in[11];
  const float* wl = (const float*)d_in[12];
  const float* gl = (const float*)d_in[13];
  const float* bl = (const float*)d_in[14];
  const float* ml = (const float*)d_in[15];
  const float* vl = (const float*)d_in[16];
  const float* wg = (const float*)d_in[17];
  const float* gg = (const float*)d_in[18];
  const float* bg = (const float*)d_in[19];
  const float* mg = (const float*)d_in[20];
  const float* vg = (const float*)d_in[21];
  const float* wp = (const float*)d_in[22];
  const float* gp = (const float*)d_in[23];
  const float* bp = (const float*)d_in[24];
  const float* mp = (const float*)d_in[25];
  const float* vp = (const float*)d_in[26];
  const float* wf = (const float*)d_in[27];
  const float* bf = (const float*)d_in[28];
  const float* lscale = (const float*)d_in[29];

  float* out = (float*)d_out;
  float* out_logits = out;
  float* out_act = out + 16 * 20;
  float* out_x = out + 16 * 20 + 16 * 20 * 1024;

  char* wsb = (char*)d_ws;
  size_t off = 0;
  auto alloc16 = [&](size_t elems) -> _Float16* {
    _Float16* p = (_Float16*)(wsb + off);
    off += ((elems * 2 + 255) & ~(size_t)255);
    return p;
  };
  _Float16* patch16 = alloc16((size_t)16384 * 768);
  _Float16* w0_16 = alloc16((size_t)512 * 768);
  _Float16* wr16 = alloc16((size_t)128 * 512);
  _Float16* wl_r = alloc16((size_t)9 * 128 * 128);
  _Float16* wg_r = alloc16((size_t)9 * 128 * 128);
  _Float16* wp16 = alloc16((size_t)512 * 256);
  _Float16* wf16 = alloc16((size_t)512 * 512);
  _Float16* pr16 = alloc16((size_t)20 * 64 * 512);
  _Float16* x1 = alloc16((size_t)16384 * 512);
  _Float16* o16 = alloc16((size_t)16384 * 128);
  _Float16* lg = alloc16((size_t)16384 * 256);
  _Float16* x2 = alloc16((size_t)16384 * 512);
  _Float16* y16 = alloc16((size_t)16384 * 512);
  _Float16* xn16 = alloc16((size_t)16384 * 512);

  auto cvt = [&](const float* s, _Float16* d, int n) {
    cvt_f16<<<(n + 255) / 256, 256, 0, stream>>>(s, d, n);
  };
  cvt(patch, patch16, 16384 * 768);
  cvt(w0, w0_16, 512 * 768);
  cvt(wr, wr16, 128 * 512);
  cvt(wp, wp16, 512 * 256);
  cvt(wf, wf16, 512 * 512);
  cvt(protos, pr16, 20 * 64 * 512);
  repack3x3<<<(9 * 128 * 128 + 255) / 256, 256, 0, stream>>>(wl, wl_r);
  repack3x3<<<(9 * 128 * 128 + 255) / 256, 256, 0, stream>>>(wg, wg_r);

  // adapter: 16384x768x512  (512x8 32x64 tiles -> 512 blocks of 8 waves)
  gemm16<768, 0><<<512, 256, 0, stream>>>(patch16, w0_16, g0, b0, m0, v0,
                                          nullptr, x1, 16384, 512, 512, 0);
  // reduce: 16384x512x128   (512x2 tiles -> 128 blocks)
  gemm16<512, 0><<<128, 256, 0, stream>>>(x1, wr16, gr, br, mr, vr, nullptr,
                                          o16, 16384, 128, 128, 0);
  // fused 3x3 convs from padded LDS slab (per-row TDM): 64 blocks, 108KB LDS
  conv_fused<<<64, 512, 12 * 36 * 128 * 2, stream>>>(
      o16, wl_r, wg_r, gl, bl, ml, vl, gg, bg, mg, vg, lg);
  // proj: 16384x256x512, BN + residual + ReLU
  gemm16<256, 1><<<512, 256, 0, stream>>>(lg, wp16, gp, bp, mp, vp, x1, x2,
                                          16384, 512, 512, 0);
  // final: 16384x512x512, +bias
  gemm16<512, 2><<<512, 256, 0, stream>>>(x2, wf16, bf, nullptr, nullptr,
                                          nullptr, nullptr, y16, 16384, 512,
                                          512, 0);
  l2norm_k<<<16384, 128, 0, stream>>>(y16, out_x, xn16);
  sims_max_k<<<2560, 256, 0, stream>>>(xn16, pr16, out_act);
  logits_k<<<320, 256, 0, stream>>>(out_act, lscale, out_logits);
}
